// Contextual_Attention_Module_6279242187257
// MI455X (gfx1250) — matrix operand order repaired and checked by reading, not compiled, not run on hardware
//
#include <hip/hip_runtime.h>

// ---------------------------------------------------------------------------
// Contextual attention (DeepFill) for MI455X / gfx1250, wave32 + WMMA f16.
// B=16, C=128, H=W=64, RATE=2, KSIZE=3, FUSE_K=3, SCALE=10.
// Heavy contractions are v_wmma_f32_16x16x32_f16 GEMMs:
//   scores : per batch  S(1024x1024) = Wn(1024x1152) * im2col(fd)(1024x1152)^T
//            (double-buffered LDS fed by global_load_async_to_lds_b128)
//   deconv : per (batch,parity) out(128x1024) over K=4096 gathered operands
//   convs  : per batch  out(128x4096) = W(128x1152) * im2col(1152x4096)
// All LDS tiles are K-contiguous per row (A: M-major, B: N-major) so both
// fragment loads are 2x ds_load_b128 per wave, no element packing.
// ---------------------------------------------------------------------------

typedef __attribute__((ext_vector_type(16))) _Float16 v16h;
typedef __attribute__((ext_vector_type(8)))  float    v8f;
typedef __attribute__((ext_vector_type(8)))  unsigned v8u;
typedef __attribute__((ext_vector_type(4)))  unsigned v4u;

#define DEV __device__ __forceinline__

// ---- WMMA fragment helpers (layouts per CDNA5 ISA) --------------

// A: 16x32 f16 tile, M-major, stride ld (f16 units, row 16B-aligned).
// lanes 0-15: M=lane, K = {0..7, 16..23}; lanes 16-31: M=lane-16, K={8..15,24..31}.
DEV v16h load_frag_a(const _Float16* t, int ld) {
  const int lane = threadIdx.x & 31;
  const int m  = lane & 15;
  const int kb = (lane & 16) >> 1;                       // 0 or 8
  const v4u lo = *(const v4u*)(t + m * ld + kb);         // K = kb .. kb+7
  const v4u hi = *(const v4u*)(t + m * ld + kb + 16);    // K = kb+16 .. kb+23
  v8u u;
#pragma unroll
  for (int i = 0; i < 4; ++i) { u[i] = lo[i]; u[4 + i] = hi[i]; }
  return __builtin_bit_cast(v16h, u);
}

// B: 32x16 f16 (K x N) stored N-major in LDS: tile[n][k], stride ld.
// lanes 0-15: N=lane, K = {0..7, 16..23}; lanes 16-31: N=lane-16, K={8..15,24..31}.
DEV v16h load_frag_b(const _Float16* t, int ld) {
  const int lane = threadIdx.x & 31;
  const int n  = lane & 15;
  const int kb = (lane & 16) >> 1;                       // 0 or 8
  const v4u lo = *(const v4u*)(t + n * ld + kb);         // K = kb .. kb+7
  const v4u hi = *(const v4u*)(t + n * ld + kb + 16);    // K = kb+16 .. kb+23
  v8u u;
#pragma unroll
  for (int i = 0; i < 4; ++i) { u[i] = lo[i]; u[4 + i] = hi[i]; }
  return __builtin_bit_cast(v16h, u);
}

DEV v8f wmma_f16(v16h a, v16h b, v8f c) {
  return __builtin_amdgcn_wmma_f32_16x16x32_f16(false, a, false, b,
                                                (short)0, c, false, false);
}

DEV v8f vzero8() { v8f z; for (int i = 0; i < 8; ++i) z[i] = 0.f; return z; }

// Wave-relative LDS byte offset of a __shared__ pointer: for shared apertures
// the low 32 bits of the flat address are the LDS address (ISA 10.2).
DEV unsigned lds_off(const void* p) {
  return (unsigned)(unsigned long long)p;
}

// Async 16-byte global -> LDS transfer (per lane), tracked by ASYNCcnt.
DEV void async_copy_b128(unsigned lds_byte, const void* gaddr) {
  asm volatile("global_load_async_to_lds_b128 %0, %1, off"
               :
               : "v"(lds_byte), "v"(gaddr)
               : "memory");
}

// ---- prep kernels ----------------------------------------------------------

__global__ void k_cast_f16(const float* __restrict__ s,
                           _Float16* __restrict__ d, int n) {
  int i = blockIdx.x * 256 + threadIdx.x;
  if (i < n) d[i] = (_Float16)s[i];
}

// 1/max(||3x3x128 patch of bd||, 1e-4) per (batch, patch)
__global__ __launch_bounds__(128) void k_inv(const float* __restrict__ bsrc,
                                             float* __restrict__ invn) {
  const int b = blockIdx.y, p = blockIdx.x;
  const int hb = p >> 5, wb = p & 31;
  const int c = threadIdx.x;
  const float* bc = bsrc + (size_t)b * 128 * 4096 + (size_t)c * 4096;
  float s = 0.f;
#pragma unroll
  for (int t = 0; t < 9; ++t) {
    int i = hb - 1 + t / 3, j = wb - 1 + t % 3;
    if (i >= 0 && i < 32 && j >= 0 && j < 32) {
      float v = bc[(2 * i) * 64 + 2 * j];      // bd = b[:, ::2, ::2]
      s += v * v;
    }
  }
  __shared__ float red[128];
  red[c] = s; __syncthreads();
  for (int st = 64; st > 0; st >>= 1) {
    if (c < st) red[c] += red[c + st];
    __syncthreads();
  }
  if (c == 0) invn[(size_t)b * 1024 + p] = 1.f / fmaxf(sqrtf(red[0]), 1e-4f);
}

// Aw[b][p][k] = f16( normalized patch of bd ), k = c*9 + kh*3 + kw
__global__ void k_packA(const float* __restrict__ bsrc,
                        const float* __restrict__ invn,
                        _Float16* __restrict__ Aw) {
  const int b = blockIdx.y;
  const int e = blockIdx.x * 256 + threadIdx.x;   // < 1024*1152
  const int p = e / 1152, k = e - p * 1152;
  const int hb = p >> 5, wb = p & 31;
  const int c = k / 9, t = k - c * 9;
  const int i = hb - 1 + t / 3, j = wb - 1 + t % 3;
  float v = 0.f;
  if (i >= 0 && i < 32 && j >= 0 && j < 32)
    v = bsrc[(size_t)b * 524288 + (size_t)c * 4096 + (2 * i) * 64 + 2 * j];
  Aw[(size_t)b * 1179648 + e] = (_Float16)(v * invn[b * 1024 + p]);
}

// Xc[b][pos][k] = f16( im2col of fd ), pos-major so score-GEMM B tiles are
// K-contiguous per row (same copy structure as A tiles).
__global__ void k_packX(const float* __restrict__ fsrc,
                        _Float16* __restrict__ Xc) {
  const int b = blockIdx.y;
  const int e = blockIdx.x * 256 + threadIdx.x;   // < 1024*1152
  const int pos = e / 1152, k = e - pos * 1152;
  const int c = k / 9, t = k - c * 9;
  const int hf = pos >> 5, wf = pos & 31;
  const int i = hf + t / 3 - 1, j = wf + t % 3 - 1;
  float v = 0.f;
  if (i >= 0 && i < 32 && j >= 0 && j < 32)
    v = fsrc[(size_t)b * 524288 + (size_t)c * 4096 + (2 * i) * 64 + 2 * j];
  Xc[(size_t)b * 1179648 + e] = (_Float16)v;
}

// ---- scores GEMM: S = Aw * Xc^T (M=N=1024, K=1152) -------------------------
// Double-buffered LDS tiles filled by async global->LDS DMA (ASYNCcnt).

__global__ __launch_bounds__(256) void k_scores(const _Float16* __restrict__ Aw,
                                                const _Float16* __restrict__ Xc,
                                                float* __restrict__ S) {
  const int batch = blockIdx.z, mblk = blockIdx.y, nblk = blockIdx.x;
  const _Float16* A = Aw + (size_t)batch * 1179648 + (size_t)mblk * 128 * 1152;
  const _Float16* Bg = Xc + (size_t)batch * 1179648 + (size_t)nblk * 128 * 1152;
  float* out = S + ((size_t)batch << 20) + (size_t)mblk * 128 * 1024 + nblk * 128;

  __shared__ alignas(16) _Float16 As[2][128 * 40];
  __shared__ alignas(16) _Float16 Bs[2][128 * 40];
  const int tid = threadIdx.x, wave = tid >> 5, lane = tid & 31;

  // This thread's two 16B chunks per 128x32 tile: chunk c -> row c>>2, col (c&3)*8
  const int c0 = tid, c1 = tid + 256;
  const int r0 = c0 >> 2, f0 = (c0 & 3) << 3;
  const int r1 = c1 >> 2, f1 = (c1 & 3) << 3;

  auto issue_tile = [&](int buf, int k0) {
    async_copy_b128(lds_off(&As[buf][r0 * 40 + f0]), A + (size_t)r0 * 1152 + k0 + f0);
    async_copy_b128(lds_off(&Bs[buf][r0 * 40 + f0]), Bg + (size_t)r0 * 1152 + k0 + f0);
    async_copy_b128(lds_off(&As[buf][r1 * 40 + f1]), A + (size_t)r1 * 1152 + k0 + f1);
    async_copy_b128(lds_off(&Bs[buf][r1 * 40 + f1]), Bg + (size_t)r1 * 1152 + k0 + f1);
  };

  v8f acc[8];
#pragma unroll
  for (int i = 0; i < 8; ++i) acc[i] = vzero8();

  issue_tile(0, 0);                                    // prologue: tile 0

  for (int kt = 0; kt < 36; ++kt) {
    const int cur = kt & 1;
    if (kt + 1 < 36) {
      __syncthreads();                                 // buf cur^1 free to overwrite
      issue_tile(cur ^ 1, (kt + 1) * 32);
      asm volatile("s_wait_asynccnt 0x4" ::: "memory");  // buf cur complete
    } else {
      asm volatile("s_wait_asynccnt 0x0" ::: "memory");
    }
    __syncthreads();                                   // buf cur visible to all waves

    v16h af = load_frag_a(&As[cur][(wave * 16) * 40], 40);
#pragma unroll
    for (int nt = 0; nt < 8; ++nt) {
      v16h bf = load_frag_b(&Bs[cur][(nt * 16) * 40], 40);
      acc[nt] = wmma_f16(af, bf, acc[nt]);
    }
  }

  const int n = lane & 15, mrow = wave * 16 + ((lane >> 4) << 3);
#pragma unroll
  for (int nt = 0; nt < 8; ++nt)
#pragma unroll
    for (int v = 0; v < 8; ++v)
      out[(size_t)(mrow + v) * 1024 + nt * 16 + n] = acc[nt][v];
}

// ---- fuse (diagonal 3-tap, twice with transpose) ---------------------------

__global__ void k_fuse1(const float* __restrict__ S, float* __restrict__ T) {
  const int b = blockIdx.y;
  const int e = blockIdx.x * 256 + threadIdx.x;       // < 1024*1024
  const int i = e >> 10, j = e & 1023;
  const float* Sb = S + ((size_t)b << 20);
  float s = 0.f;
#pragma unroll
  for (int d = -1; d <= 1; ++d) {
    int ii = i + d, jj = j + d;
    if (ii >= 0 && ii < 1024 && jj >= 0 && jj < 1024)
      s += Sb[(size_t)ii * 1024 + jj];
  }
  T[((size_t)b << 20) + e] = s;
}

__global__ void k_fuse2(const float* __restrict__ T, float* __restrict__ F) {
  const int b = blockIdx.y;
  const int e = blockIdx.x * 256 + threadIdx.x;
  const int i = e >> 10, j = e & 1023;                 // i=hb*32+wb, j=hf*32+wf
  const int hb = i >> 5, wb = i & 31, hf = j >> 5, wf = j & 31;
  const int a0 = wb * 32 + hb, b0 = wf * 32 + hf;      // transposed flat index
  const float* Tb = T + ((size_t)b << 20);
  float s = 0.f;
#pragma unroll
  for (int d = -1; d <= 1; ++d) {
    int a = a0 + d, bb = b0 + d;
    if (a >= 0 && a < 1024 && bb >= 0 && bb < 1024) {
      int ai = (a & 31) * 32 + (a >> 5);               // map back to (hb,wb)
      int bi = (bb & 31) * 32 + (bb >> 5);
      s += Tb[(size_t)ai * 1024 + bi];
    }
  }
  F[((size_t)b << 20) + e] = s;
}

// ---- softmax over the 1024 patches (axis p), scale 10, f16 output ----------

__global__ __launch_bounds__(256) void k_softmax(const float* __restrict__ F,
                                                 _Float16* __restrict__ attn) {
  const int b = blockIdx.y, pos = blockIdx.x, tid = threadIdx.x;
  const float* col = F + ((size_t)b << 20) + pos;
  _Float16* o = attn + ((size_t)b << 20) + pos;
  __shared__ float red[256];
  float v[4], mx = -1e30f;
#pragma unroll
  for (int q = 0; q < 4; ++q) {
    v[q] = col[(size_t)(tid + 256 * q) * 1024] * 10.0f;
    mx = fmaxf(mx, v[q]);
  }
  red[tid] = mx; __syncthreads();
  for (int s = 128; s > 0; s >>= 1) {
    if (tid < s) red[tid] = fmaxf(red[tid], red[tid + s]);
    __syncthreads();
  }
  mx = red[0]; __syncthreads();
  float sm = 0.f;
#pragma unroll
  for (int q = 0; q < 4; ++q) { v[q] = __expf(v[q] - mx); sm += v[q]; }
  red[tid] = sm; __syncthreads();
  for (int s = 128; s > 0; s >>= 1) {
    if (tid < s) red[tid] += red[tid + s];
    __syncthreads();
  }
  const float inv = 1.f / red[0];
#pragma unroll
  for (int q = 0; q < 4; ++q)
    o[(size_t)(tid + 256 * q) * 1024] = (_Float16)(v[q] * inv);
}

// ---- transposed-conv paste as 4 parity-class GEMMs (K=4096) ----------------
// out[co, 2yy+py, 2xx+px] = 1/4 * sum_{khi,kwi,p} b[co,2hb+kh-1,2wb+kw-1] *
//                           attn[p, (yy-khi+py)*32 + (xx-kwi+px)]
// kh = 2*khi + (1-py), kw = 2*kwi + (1-px), p = hb*32+wb.

__global__ __launch_bounds__(256) void k_deconv(const _Float16* __restrict__ bh,
                                                const _Float16* __restrict__ attn,
                                                float* __restrict__ Y) {
  const int batch = blockIdx.z;
  const int cls = blockIdx.y, py = cls >> 1, px = cls & 1;
  const int nblk = blockIdx.x;                         // 64 positions per block
  const _Float16* bb = bh + (size_t)batch * 524288;
  const _Float16* at = attn + ((size_t)batch << 20);
  float* Yo = Y + (size_t)batch * 524288;

  __shared__ alignas(16) _Float16 As[128 * 40];
  __shared__ alignas(16) _Float16 Bs[64 * 40];         // N-major: [n][k]
  const int tid = threadIdx.x, wave = tid >> 5, lane = tid & 31;

  v8f acc[4];
#pragma unroll
  for (int i = 0; i < 4; ++i) acc[i] = vzero8();

  for (int k0 = 0; k0 < 4096; k0 += 32) {
    __syncthreads();
    for (int e = tid; e < 4096; e += 256) {            // A gather from b (f16)
      int co = e >> 5, kk = e & 31, kc = k0 + kk;
      int tap = kc >> 10, khi = tap >> 1, kwi = tap & 1;
      int p = kc & 1023, hb = p >> 5, wb = p & 31;
      int r = 2 * hb + 2 * khi - py;                   // 2hb + kh - 1
      int c2 = 2 * wb + 2 * kwi - px;
      _Float16 v = (_Float16)0.f;
      if (r >= 0 && r < 64 && c2 >= 0 && c2 < 64)
        v = bb[(size_t)co * 4096 + r * 64 + c2];
      As[co * 40 + kk] = v;
    }
    for (int e = tid; e < 2048; e += 256) {            // B gather from attn
      int kk = e >> 6, nn = e & 63, kc = k0 + kk;
      int tap = kc >> 10, khi = tap >> 1, kwi = tap & 1;
      int p = kc & 1023;
      int n = nblk * 64 + nn, yy = n >> 5, xx = n & 31;
      int i = yy - khi + py, j = xx - kwi + px;
      _Float16 v = (_Float16)0.f;
      if (i >= 0 && i < 32 && j >= 0 && j < 32)
        v = at[(size_t)p * 1024 + i * 32 + j];
      Bs[nn * 40 + kk] = v;                            // N-major store
    }
    __syncthreads();
    v16h af = load_frag_a(&As[(wave * 16) * 40], 40);
#pragma unroll
    for (int nt = 0; nt < 4; ++nt) {
      v16h bf = load_frag_b(&Bs[(nt * 16) * 40], 40);
      acc[nt] = wmma_f16(af, bf, acc[nt]);
    }
  }
  const int nl = lane & 15, mrow = wave * 16 + ((lane >> 4) << 3);
#pragma unroll
  for (int nt = 0; nt < 4; ++nt)
#pragma unroll
    for (int v = 0; v < 8; ++v) {
      int m = mrow + v;
      int n = nblk * 64 + nt * 16 + nl, yy = n >> 5, xx = n & 31;
      Yo[(size_t)m * 4096 + (2 * yy + py) * 64 + (2 * xx + px)] =
          0.25f * acc[nt][v];
    }
}

// ---- 3x3 conv (C=128->128) as implicit GEMM + bias + ELU -------------------

__global__ __launch_bounds__(256) void k_conv_elu(const float* __restrict__ x,
                                                  const _Float16* __restrict__ Wm,
                                                  const float* __restrict__ bias,
                                                  float* __restrict__ out) {
  const int batch = blockIdx.y, nblk = blockIdx.x;     // 64 pixels per block
  const float* xin = x + (size_t)batch * 524288;
  float* o = out + (size_t)batch * 524288 + nblk * 64;

  __shared__ alignas(16) _Float16 As[128 * 40];
  __shared__ alignas(16) _Float16 Bs[64 * 40];         // N-major: [n][k]
  const int tid = threadIdx.x, wave = tid >> 5, lane = tid & 31;

  v8f acc[4];
#pragma unroll
  for (int i = 0; i < 4; ++i) acc[i] = vzero8();

  for (int k0 = 0; k0 < 1152; k0 += 32) {
    __syncthreads();
#pragma unroll
    for (int c = tid; c < 1024; c += 256) {            // A: weights 128x32
      int r = c >> 3, cc = (c & 7) << 2;
      *(unsigned long long*)&As[r * 40 + cc] =
          *(const unsigned long long*)&Wm[(size_t)r * 1152 + k0 + cc];
    }
    for (int e = tid; e < 2048; e += 256) {            // B: im2col 32x64
      int kk = e >> 6, nn = e & 63, k = k0 + kk;
      int ci = k / 9, t = k - ci * 9;
      int dh = t / 3 - 1, dw = t % 3 - 1;
      int n = nblk * 64 + nn, y = n >> 6, xq = n & 63;
      int yy = y + dh, xx = xq + dw;
      float v = 0.f;
      if (yy >= 0 && yy < 64 && xx >= 0 && xx < 64)
        v = xin[(size_t)ci * 4096 + yy * 64 + xx];
      Bs[nn * 40 + kk] = (_Float16)v;                  // N-major store
    }
    __syncthreads();
    v16h af = load_frag_a(&As[(wave * 16) * 40], 40);
#pragma unroll
    for (int nt = 0; nt < 4; ++nt) {
      v16h bf = load_frag_b(&Bs[(nt * 16) * 40], 40);
      acc[nt] = wmma_f16(af, bf, acc[nt]);
    }
  }
  const int nl = lane & 15, mrow = wave * 16 + ((lane >> 4) << 3);
#pragma unroll
  for (int nt = 0; nt < 4; ++nt)
#pragma unroll
    for (int v = 0; v < 8; ++v) {
      int m = mrow + v;
      float r = acc[nt][v] + bias[m];
      r = r > 0.f ? r : (__expf(r) - 1.f);             // ELU
      o[(size_t)m * 4096 + nt * 16 + nl] = r;
    }
}

// ---- host launcher ---------------------------------------------------------

extern "C" void kernel_launch(void* const* d_in, const int* in_sizes, int n_in,
                              void* d_out, int out_size, void* d_ws,
                              size_t ws_size, hipStream_t stream) {
  const float* f_in = (const float*)d_in[0];
  const float* b_in = (const float*)d_in[1];
  const float* W1 = (const float*)d_in[2];
  const float* b1 = (const float*)d_in[3];
  const float* W2 = (const float*)d_in[4];
  const float* b2 = (const float*)d_in[5];
  float* out = (float*)d_out;

  char* ws = (char*)d_ws;
  size_t off = 0;
  auto carve = [&](size_t bytes) {
    char* p = ws + off;
    off += (bytes + 255) & ~(size_t)255;
    return p;
  };
  _Float16* Aw  = (_Float16*)carve(16ULL * 1024 * 1152 * 2);  // patch weights
  _Float16* Xc  = (_Float16*)carve(16ULL * 1024 * 1152 * 2);  // im2col(fd), pos-major
  float*    S   = (float*)   carve(16ULL * 1024 * 1024 * 4);  // scores / fused
  float*    T1  = (float*)   carve(16ULL * 1024 * 1024 * 4);  // fuse temp
  float*    inv = (float*)   carve(16ULL * 1024 * 4);         // 1/patch-norm
  _Float16* bh  = (_Float16*)carve(16ULL * 128 * 4096 * 2);   // f16(b)
  float*    Y   = (float*)   carve(16ULL * 128 * 4096 * 4);   // attention out
  float*    Hb  = (float*)   carve(16ULL * 128 * 4096 * 4);   // conv1 out
  _Float16* Wm1 = (_Float16*)carve(128ULL * 1152 * 2);
  _Float16* Wm2 = (_Float16*)carve(128ULL * 1152 * 2);
  _Float16* attn = Aw;  // Aw is dead after k_scores; reuse for softmax output

  k_cast_f16<<<dim3(32768), 256, 0, stream>>>(b_in, bh, 16 * 128 * 4096);
  k_cast_f16<<<dim3(576), 256, 0, stream>>>(W1, Wm1, 128 * 1152);
  k_cast_f16<<<dim3(576), 256, 0, stream>>>(W2, Wm2, 128 * 1152);
  k_inv<<<dim3(1024, 16), 128, 0, stream>>>(b_in, inv);
  k_packA<<<dim3(4608, 16), 256, 0, stream>>>(b_in, inv, Aw);
  k_packX<<<dim3(4608, 16), 256, 0, stream>>>(f_in, Xc);
  k_scores<<<dim3(8, 8, 16), 256, 0, stream>>>(Aw, Xc, S);
  k_fuse1<<<dim3(4096, 16), 256, 0, stream>>>(S, T1);
  k_fuse2<<<dim3(4096, 16), 256, 0, stream>>>(T1, S);
  k_softmax<<<dim3(1024, 16), 256, 0, stream>>>(S, attn);
  k_deconv<<<dim3(16, 4, 16), 256, 0, stream>>>(bh, attn, Y);
  k_conv_elu<<<dim3(64, 16), 256, 0, stream>>>(Y, Wm1, b1, Hb);
  k_conv_elu<<<dim3(64, 16), 256, 0, stream>>>(Hb, Wm2, b2, out);
}
